// LSTMDT_58548994179401
// MI455X (gfx1250) — matrix operand order repaired and checked by reading, not compiled, not run on hardware
//
#include <hip/hip_runtime.h>
#include <hip/hip_bf16.h>

#define Bsz 256
#define Tsz 128
#define Dsz 256
#define Hsz 512
#define H4  (4*Hsz)
#define Mrows (Bsz*Tsz)

typedef __attribute__((ext_vector_type(16))) _Float16 v16h;
typedef __attribute__((ext_vector_type(8)))  float    v8f;

__device__ __forceinline__ float sigmoidf_(float x) { return 1.0f / (1.0f + __expf(-x)); }

// ---------------------------------------------------------------------------
// K0a: bulk f32 -> f16 conversion of dt ([B,T,D], 8.4M elements)
// ---------------------------------------------------------------------------
__global__ void k_cvt_dt(const float* __restrict__ dt, _Float16* __restrict__ dt16) {
  const int idx = blockIdx.x * blockDim.x + threadIdx.x;
  if (idx < Mrows * Dsz) dt16[idx] = (_Float16)dt[idx];
}

// ---------------------------------------------------------------------------
// K0b: one-time prep.
//  - W_ih [4H,D], W_hh [4H,H]  -> f16 (already [N,K] layout for B tiles)
//  - w_gx [D,D] -> f16 transposed [N=D, K=D]
//  - w_gh [D,H] -> f16 transposed [N=H, K=D]
//  - broadcast h0/c0 into f32 state buffers
//  - bias = b_ih + b_hh
// ---------------------------------------------------------------------------
__global__ void k_prep(const float* __restrict__ W_ih, const float* __restrict__ W_hh,
                       const float* __restrict__ w_gx, const float* __restrict__ w_gh,
                       const float* __restrict__ h0, const float* __restrict__ c0,
                       const float* __restrict__ b_ih, const float* __restrict__ b_hh,
                       _Float16* __restrict__ Wih_h, _Float16* __restrict__ Whh_h,
                       _Float16* __restrict__ wgxT_h, _Float16* __restrict__ wghT_h,
                       float* __restrict__ hbuf, float* __restrict__ cbuf0,
                       float* __restrict__ bias) {
  const int n0 = H4 * Dsz;                    // Wih
  const int n1 = n0 + H4 * Hsz;               // Whh
  const int n2 = n1 + Dsz * Dsz;              // wgxT
  const int n3 = n2 + Dsz * Hsz;              // wghT
  const int n4 = n3 + Bsz * Hsz;              // h/c init
  const int n5 = n4 + H4;                     // bias
  for (int idx = blockIdx.x * blockDim.x + threadIdx.x; idx < n5;
       idx += gridDim.x * blockDim.x) {
    if (idx < n0) {
      Wih_h[idx] = (_Float16)W_ih[idx];
    } else if (idx < n1) {
      int j = idx - n0;
      Whh_h[j] = (_Float16)W_hh[j];
    } else if (idx < n2) {
      int j = idx - n1;                       // j = k*D + n in source
      int k = j >> 8, n = j & (Dsz - 1);
      wgxT_h[(size_t)n * Dsz + k] = (_Float16)w_gx[j];
    } else if (idx < n3) {
      int j = idx - n2;                       // j = k*H + n in source
      int k = j >> 9, n = j & (Hsz - 1);
      wghT_h[(size_t)n * Dsz + k] = (_Float16)w_gh[j];
    } else if (idx < n4) {
      int j = idx - n3;
      int col = j & (Hsz - 1);
      hbuf[j]  = h0[col];
      cbuf0[j] = c0[col];
    } else {
      int j = idx - n4;
      bias[j] = b_ih[j] + b_hh[j];
    }
  }
}

// ---------------------------------------------------------------------------
// K1: all-timesteps precompute. Rows m = b*T + t, M = 32768.
// Columns [0,D)   : gamma_x -> xs epilogue (xhat output + f16 copy)
// Columns [D,D+H) : gamma_h -> workspace
// One wave per 16x16 tile; K = D = 256 -> 8 WMMAs per tile. All-f16 operands,
// all contiguous b128 loads.
// ---------------------------------------------------------------------------
__global__ void k_precompute(const float* __restrict__ x, const _Float16* __restrict__ dt16,
                             const float* __restrict__ mask, const float* __restrict__ x_mean,
                             const _Float16* __restrict__ wgxT_h, const float* __restrict__ h_gx,
                             const _Float16* __restrict__ wghT_h, const float* __restrict__ h_gh,
                             _Float16* __restrict__ xs_h, float* __restrict__ gammaH,
                             float* __restrict__ xhat_out) {
  const int lane = threadIdx.x & 31;
  const int wave = threadIdx.x >> 5;
  const int NT = (Dsz + Hsz) / 16;                  // 48 column tiles
  const int tid = blockIdx.x * 4 + wave;
  const int tile_m = tid / NT;
  const int tile_n = tid % NT;
  const int row0 = tile_m * 16;
  const int n0   = tile_n * 16;

  const int arow  = row0 + (lane & 15);
  const int akoff = (lane >> 4) << 4;
  const int ncol  = lane & 15;
  const int bkoff = (lane >> 4) << 4;
  const int rr0   = (lane >> 4) << 3;

  const _Float16* wT = (n0 < Dsz) ? wgxT_h : wghT_h;
  const int nbase = (n0 < Dsz) ? (n0 + ncol) : (n0 - Dsz + ncol);

  v8f acc = {};
  for (int k0 = 0; k0 < Dsz; k0 += 32) {
    const _Float16* ap = dt16 + (size_t)arow * Dsz + k0 + akoff;
    v16h a;
#pragma unroll
    for (int i = 0; i < 16; ++i) a[i] = ap[i];
    const _Float16* bp = wT + (size_t)nbase * Dsz + k0 + bkoff;
    v16h bm;
#pragma unroll
    for (int i = 0; i < 16; ++i) bm[i] = bp[i];
    acc = __builtin_amdgcn_wmma_f32_16x16x32_f16(false, a, false, bm, (short)0, acc, false, false);
  }

  if (n0 < Dsz) {
    const int d = nbase;
    const float xm = x_mean[d];
    const float hg = h_gx[d];
#pragma unroll
    for (int r = 0; r < 8; ++r) {
      const int m  = row0 + rr0 + r;
      const int t  = m & (Tsz - 1);
      const int b_ = m >> 7;                        // T = 128
      const float g = __expf(-fmaxf(0.0f, acc[r] + hg));
      float imput;
      if (t == 0) {
        imput = xm;
      } else {
        const float xp = x[(size_t)(m - 1) * Dsz + d];
        imput = g * xp + (1.0f - g) * xm;
      }
      const float xv = x[(size_t)m * Dsz + d];
      const float mk = mask[(size_t)m * Dsz + d];
      const float xs = xv * mk + (1.0f - mk) * imput;
      xhat_out[((size_t)t * Bsz + b_) * Dsz + d] = xs;          // output [T,B,D]
      xs_h[(size_t)m * Dsz + d] = (_Float16)xs;                 // f16 for GEMM A
    }
  } else {
    const int hc = nbase;
    const float hg = h_gh[hc];
#pragma unroll
    for (int r = 0; r < 8; ++r) {
      const int m = row0 + rr0 + r;
      gammaH[(size_t)m * Hsz + hc] = __expf(-fmaxf(0.0f, acc[r] + hg));
    }
  }
}

// ---------------------------------------------------------------------------
// K1b: seed hdec16 buffer 0:  hdec16[b,j] = (f16)(gammaH[b,t=0,j] * h0[j])
// ---------------------------------------------------------------------------
__global__ void k_decay0(const float* __restrict__ gammaH, const float* __restrict__ h0,
                         _Float16* __restrict__ hdec0) {
  const int idx = blockIdx.x * blockDim.x + threadIdx.x;
  if (idx >= Bsz * Hsz) return;
  const int b_ = idx >> 9, j = idx & (Hsz - 1);
  hdec0[idx] = (_Float16)(gammaH[(size_t)b_ * Tsz * Hsz + j] * h0[j]);
}

// ---------------------------------------------------------------------------
// K2 (per timestep): gates = xs_t @ W_ih^T + hdec16 @ W_hh^T + bias
// 4 waves/block compute the i/f/g/o 16x16 tiles at the same (rows, j) spot,
// exchange via LDS, then fused LSTM update. All GEMM operands are contiguous
// f16 loads. Epilogue also writes NEXT step's decayed hidden state in f16.
// ---------------------------------------------------------------------------
__global__ void k_step(int t,
                       const _Float16* __restrict__ xs_h,
                       const _Float16* __restrict__ Wih_h,
                       const _Float16* __restrict__ Whh_h,
                       const float* __restrict__ gammaH,
                       const float* __restrict__ bias,
                       const float* __restrict__ tp,
                       const float* __restrict__ w_t, const float* __restrict__ h_t,
                       const _Float16* __restrict__ hdec_prev,
                       _Float16* __restrict__ hdec_next,
                       const float* __restrict__ c_prev, float* __restrict__ c_next,
                       float* __restrict__ h_out) {
  __shared__ float smem[4 * 16 * 16];
  const int lane = threadIdx.x & 31;
  const int wave = threadIdx.x >> 5;
  const int tileRow = blockIdx.x & 15;              // 16 row tiles (B=256)
  const int tileJ   = blockIdx.x >> 4;              // 32 j tiles  (H=512)
  const int row0 = tileRow * 16;
  const int j0   = tileJ * 16;
  const int gc0  = wave * Hsz + j0;                 // this wave's gate column base

  const int brow  = row0 + (lane & 15);
  const int akoff = (lane >> 4) << 4;
  const int ncol  = lane & 15;
  const int bkoff = (lane >> 4) << 4;
  const int rr0   = (lane >> 4) << 3;

  v8f acc = {};
  // input contribution: xs[m = brow*T + t] @ W_ih^T   (K = D = 256)
  for (int k0 = 0; k0 < Dsz; k0 += 32) {
    const _Float16* ap = xs_h + ((size_t)brow * Tsz + t) * Dsz + k0 + akoff;
    v16h a;
#pragma unroll
    for (int i = 0; i < 16; ++i) a[i] = ap[i];
    const _Float16* bp = Wih_h + (size_t)(gc0 + ncol) * Dsz + k0 + bkoff;
    v16h bm;
#pragma unroll
    for (int i = 0; i < 16; ++i) bm[i] = bp[i];
    acc = __builtin_amdgcn_wmma_f32_16x16x32_f16(false, a, false, bm, (short)0, acc, false, false);
  }
  // recurrent contribution: hdec16 @ W_hh^T   (K = H = 512)
  for (int k0 = 0; k0 < Hsz; k0 += 32) {
    const _Float16* ap = hdec_prev + (size_t)brow * Hsz + k0 + akoff;
    v16h a;
#pragma unroll
    for (int i = 0; i < 16; ++i) a[i] = ap[i];
    const _Float16* bp = Whh_h + (size_t)(gc0 + ncol) * Hsz + k0 + bkoff;
    v16h bm;
#pragma unroll
    for (int i = 0; i < 16; ++i) bm[i] = bp[i];
    acc = __builtin_amdgcn_wmma_f32_16x16x32_f16(false, a, false, bm, (short)0, acc, false, false);
  }

  const float bs = bias[gc0 + ncol];
#pragma unroll
  for (int r = 0; r < 8; ++r)
    smem[wave * 256 + (rr0 + r) * 16 + ncol] = acc[r] + bs;
  __syncthreads();

  // fused LSTM update on the 16x16 (b, j) tile
  for (int e = threadIdx.x; e < 256; e += 128) {
    const int rr = e >> 4, jj = e & 15;
    const int b_ = row0 + rr;
    const int j  = j0 + jj;
    const float ig = smem[0 * 256 + e];
    const float fg = smem[1 * 256 + e];
    const float gg = smem[2 * 256 + e];
    const float og = smem[3 * 256 + e];
    const float gH   = gammaH[((size_t)b_ * Tsz + t) * Hsz + j];
    const float cdec = gH * c_prev[(size_t)b_ * Hsz + j];
    const float c1 = sigmoidf_(fg) * cdec + sigmoidf_(ig) * tanhf(gg);
    const float h1 = sigmoidf_(og) * tanhf(c1);
    const float gt = sigmoidf_(tp[(size_t)b_ * Tsz + t] * w_t[j] + h_t[j]);
    const float hnew = gt * h1;
    h_out[(size_t)b_ * Hsz + j] = hnew;
    c_next[(size_t)b_ * Hsz + j] = gt * c1;
    if (t + 1 < Tsz) {
      // pre-decay the hidden state for the NEXT step in f16
      const float gHn = gammaH[((size_t)b_ * Tsz + (t + 1)) * Hsz + j];
      hdec_next[(size_t)b_ * Hsz + j] = (_Float16)(gHn * hnew);
    }
  }
}

// ---------------------------------------------------------------------------
// K3 (per timestep): output head. One thread per batch row.
// ---------------------------------------------------------------------------
__global__ void k_head(int t, const float* __restrict__ h_new,
                       const float* __restrict__ out_W1, const float* __restrict__ out_b1,
                       const float* __restrict__ out_W2, const float* __restrict__ out_b2,
                       float* __restrict__ outs) {
  const int b_ = blockIdx.x * blockDim.x + threadIdx.x;
  if (b_ >= Bsz) return;
  const float* hr = h_new + (size_t)b_ * Hsz;
  float z1[10];
#pragma unroll
  for (int q = 0; q < 10; ++q) z1[q] = out_b1[q];
  for (int j = 0; j < Hsz; ++j) {
    const float hv = hr[j];
#pragma unroll
    for (int q = 0; q < 10; ++q) z1[q] += hv * out_W1[j * 10 + q];
  }
#pragma unroll
  for (int q = 0; q < 10; ++q) z1[q] = sigmoidf_(z1[q]);
  float z2[2];
#pragma unroll
  for (int r = 0; r < 2; ++r) {
    float s = out_b2[r];
#pragma unroll
    for (int q = 0; q < 10; ++q) s += z1[q] * out_W2[q * 2 + r];
    z2[r] = sigmoidf_(s);
  }
  const float mx = fmaxf(z2[0], z2[1]);
  const float e0 = __expf(z2[0] - mx), e1 = __expf(z2[1] - mx);
  const float inv = 1.0f / (e0 + e1);
  outs[((size_t)t * Bsz + b_) * 2 + 0] = e0 * inv;
  outs[((size_t)t * Bsz + b_) * 2 + 1] = e1 * inv;
}

// ---------------------------------------------------------------------------
extern "C" void kernel_launch(void* const* d_in, const int* in_sizes, int n_in,
                              void* d_out, int out_size, void* d_ws, size_t ws_size,
                              hipStream_t stream) {
  (void)in_sizes; (void)n_in; (void)out_size; (void)ws_size;
  const float* x      = (const float*)d_in[0];
  const float* dt     = (const float*)d_in[1];
  const float* mask   = (const float*)d_in[2];
  const float* tp     = (const float*)d_in[3];
  const float* x_mean = (const float*)d_in[4];
  const float* h0     = (const float*)d_in[5];
  const float* c0     = (const float*)d_in[6];
  const float* w_gx   = (const float*)d_in[7];
  const float* h_gx   = (const float*)d_in[8];
  const float* w_gh   = (const float*)d_in[9];
  const float* h_gh   = (const float*)d_in[10];
  const float* w_t    = (const float*)d_in[11];
  const float* h_t    = (const float*)d_in[12];
  const float* W_ih   = (const float*)d_in[13];
  const float* W_hh   = (const float*)d_in[14];
  const float* b_ih   = (const float*)d_in[15];
  const float* b_hh   = (const float*)d_in[16];
  const float* out_W1 = (const float*)d_in[17];
  const float* out_b1 = (const float*)d_in[18];
  const float* out_W2 = (const float*)d_in[19];
  const float* out_b2 = (const float*)d_in[20];

  float* outs = (float*)d_out;
  float* xhat = outs + (size_t)Tsz * Bsz * 2;

  // workspace layout (~107 MB)
  char* w = (char*)d_ws;
  size_t o = 0;
  _Float16* Wih_h  = (_Float16*)(w + o); o += (size_t)H4 * Dsz * 2;       //  1.0 MB
  _Float16* Whh_h  = (_Float16*)(w + o); o += (size_t)H4 * Hsz * 2;       //  2.0 MB
  _Float16* wgxT_h = (_Float16*)(w + o); o += (size_t)Dsz * Dsz * 2;      //  0.13 MB
  _Float16* wghT_h = (_Float16*)(w + o); o += (size_t)Hsz * Dsz * 2;      //  0.26 MB
  _Float16* dt16   = (_Float16*)(w + o); o += (size_t)Mrows * Dsz * 2;    // 16.8 MB
  _Float16* xs_h   = (_Float16*)(w + o); o += (size_t)Mrows * Dsz * 2;    // 16.8 MB
  float* gammaH    = (float*)(w + o);    o += (size_t)Mrows * Hsz * 4;    // 67.1 MB
  float* hbuf      = (float*)(w + o);    o += (size_t)Bsz * Hsz * 4;      //  0.5 MB
  float* cb        = (float*)(w + o);    o += (size_t)2 * Bsz * Hsz * 4;  //  1.0 MB
  _Float16* hdec   = (_Float16*)(w + o); o += (size_t)2 * Bsz * Hsz * 2;  //  0.5 MB
  float* bias      = (float*)(w + o);                                     //  8 KB

  // K0a: dt -> f16 (one-time, removes per-tile cvt chains in K1)
  k_cvt_dt<<<(Mrows * Dsz + 255) / 256, 256, 0, stream>>>(dt, dt16);

  // K0b: weight conversion/transposition + state init
  {
    const int n5 = H4 * Dsz + H4 * Hsz + Dsz * Dsz + Dsz * Hsz + Bsz * Hsz + H4;
    k_prep<<<(n5 + 255) / 256, 256, 0, stream>>>(W_ih, W_hh, w_gx, w_gh, h0, c0,
                                                 b_ih, b_hh, Wih_h, Whh_h,
                                                 wgxT_h, wghT_h, hbuf, cb, bias);
  }

  // K1: all-timestep gamma_x/xs/gamma_h (98304 tiles, 4 waves/block)
  {
    const int tiles = (Mrows / 16) * ((Dsz + Hsz) / 16);
    k_precompute<<<tiles / 4, 128, 0, stream>>>(x, dt16, mask, x_mean, wgxT_h, h_gx,
                                                wghT_h, h_gh, xs_h, gammaH, xhat);
  }

  // K1b: seed decayed hidden state for t=0
  k_decay0<<<(Bsz * Hsz + 255) / 256, 256, 0, stream>>>(gammaH, h0, hdec);

  // sequential recurrence
  for (int t = 0; t < Tsz; ++t) {
    _Float16* hdec_prev = hdec + (size_t)(t & 1) * Bsz * Hsz;
    _Float16* hdec_next = hdec + (size_t)((t + 1) & 1) * Bsz * Hsz;
    float* c_prev = cb + (size_t)(t & 1) * Bsz * Hsz;
    float* c_next = cb + (size_t)((t + 1) & 1) * Bsz * Hsz;
    k_step<<<16 * 32, 128, 0, stream>>>(t, xs_h, Wih_h, Whh_h, gammaH, bias,
                                        tp, w_t, h_t, hdec_prev, hdec_next,
                                        c_prev, c_next, hbuf);
    k_head<<<2, 128, 0, stream>>>(t, hbuf, out_W1, out_b1, out_W2, out_b2, outs);
  }
}
